// ImprovedBindingField_38843684225999
// MI455X (gfx1250) — matrix operand order repaired and checked by reading, not compiled, not run on hardware
//
#include <hip/hip_runtime.h>
#include <math.h>

// ---------------------------------------------------------------------------
// Problem constants (B=8, T=4096, D=512)
// ---------------------------------------------------------------------------
#define BSZ   8
#define TSEQ  4096
#define DIM   512
#define NPOS  4095              // output positions per batch
#define NPAT  100               // patterns
#define NPATP 112               // patterns padded to multiple of 16
#define MTILE 32                // positions per workgroup (two 16-row M tiles)
#define XROWS 34                // rows of x staged per tile (p0-1 .. p0+32)
#define XSTRIDE 528             // bf16 elems per LDS row (512 + 16 pad)

// output offsets (flat concat in return order)
#define OFF_EB (BSZ * TSEQ)                 // 32768
#define OFF_AT (OFF_EB + BSZ * NPOS)        // 65528
#define OFF_BS (OFF_AT + BSZ * NPOS)        // 98288
#define OFF_CM (OFF_BS + BSZ * NPOS)        // 131048
#define OFF_PB (OFF_CM + BSZ * NPOS)        // 163808

typedef __attribute__((ext_vector_type(16))) __bf16 v16bf;
typedef __attribute__((ext_vector_type(8)))  float  v8f;

union FragB { v16bf v; uint4 q[2]; };

// ---------------------------------------------------------------------------
// Fragment loaders
// A (16x32 bf16): lane L holds M = L&15 ; K-chunks [kt*32 + (L>>4)*8 .. +7]
// and [kt*32 + 16 + (L>>4)*8 .. +7]  -> two 16B LDS loads.
// B (32x16 bf16): packed so each lane reads 32 contiguous bytes.
// ---------------------------------------------------------------------------
__device__ __forceinline__ v16bf load_a_frag(const __bf16* xrow, int kt, int lh) {
  const __bf16* base = xrow + kt * 32 + lh * 8;
  FragB f;
  f.q[0] = *(const uint4*)(base);
  f.q[1] = *(const uint4*)(base + 16);
  return f.v;
}

__device__ __forceinline__ v16bf load_b_frag(const __bf16* pw, int Nt, int kt,
                                             int nt, int lane) {
  const __bf16* base = pw + ((size_t)((kt * Nt + nt) * 32 + lane) << 4);
  FragB f;
  f.q[0] = ((const uint4*)base)[0];
  f.q[1] = ((const uint4*)base)[1];
  return f.v;
}

// 32x16 output tile (two 16x16 WMMA tiles sharing every B fragment), K = 512.
// Two independent accumulator chains -> ILP-2 on the XDL pipe and half the
// B-matrix L2 traffic vs. a 16-row tile.
__device__ __forceinline__ void gemm32(const __bf16* xs, int row_off,
                                       const __bf16* pw, int Nt, int nt,
                                       int lane, v8f& c0, v8f& c1) {
  const int lh = lane >> 4;
  const __bf16* xrow0 = xs + (row_off + (lane & 15)) * XSTRIDE;
  const __bf16* xrow1 = xrow0 + 16 * XSTRIDE;
#pragma unroll 4
  for (int kt = 0; kt < 16; ++kt) {
    v16bf b  = load_b_frag(pw, Nt, kt, nt, lane);
    v16bf a0 = load_a_frag(xrow0, kt, lh);
    v16bf a1 = load_a_frag(xrow1, kt, lh);
    c0 = __builtin_amdgcn_wmma_f32_16x16x32_bf16(false, a0, false, b,
                                                 (short)0, c0, false, false);
    c1 = __builtin_amdgcn_wmma_f32_16x16x32_bf16(false, a1, false, b,
                                                 (short)0, c1, false, false);
  }
}

__device__ __forceinline__ float sigm(float x) {
  return 1.0f / (1.0f + __expf(-x));
}

// ---------------------------------------------------------------------------
// Prep kernels (run each launch; cheap vs. the 91 GFLOP main pass)
// ---------------------------------------------------------------------------
__global__ void norm_patterns_kernel(const float* __restrict__ pm,
                                     float* __restrict__ Pn) {
  __shared__ float red[256];
  const int n = blockIdx.x;
  const float* row = pm + n * DIM;
  float s = 0.f;
  for (int d = threadIdx.x; d < DIM; d += 256) { float v = row[d]; s += v * v; }
  red[threadIdx.x] = s;
  __syncthreads();
  for (int off = 128; off > 0; off >>= 1) {
    if (threadIdx.x < off) red[threadIdx.x] += red[threadIdx.x + off];
    __syncthreads();
  }
  const float inv = 1.0f / fmaxf(sqrtf(red[0]), 1e-12f);
  for (int d = threadIdx.x; d < DIM; d += 256) Pn[n * DIM + d] = row[d] * inv;
}

// Wps[k][n] = sum_d pk_W[k][d] * Pn[n][d]   (512 x 112, zero-padded cols)
__global__ void wps_kernel(const float* __restrict__ pkW,
                           const float* __restrict__ Pn,
                           float* __restrict__ Wps) {
  const int tid = blockIdx.x * 256 + threadIdx.x;
  if (tid >= DIM * NPATP) return;
  const int k = tid / NPATP, n = tid - k * NPATP;
  float s = 0.f;
  if (n < NPAT) {
    const float* a = pkW + k * DIM;
    const float* b = Pn + n * DIM;
    for (int d = 0; d < DIM; ++d) s += a[d] * b[d];
  }
  Wps[k * NPATP + n] = s;
}

__global__ void bps_kernel(const float* __restrict__ pk_b,
                           const float* __restrict__ Pn,
                           float* __restrict__ bps) {
  const int n = threadIdx.x;
  if (n >= NPATP) return;
  float s = 0.f;
  if (n < NPAT)
    for (int d = 0; d < DIM; ++d) s += pk_b[d] * Pn[n * DIM + d];
  bps[n] = s;
}

// Repack a row-major f32 KxN weight into the 32-lane bf16 B-fragment layout:
// dst[((kt*Nt+nt)*32 + lane)*16 + e] = W[kt*32 + (lane>>4)*8 + (e&7) + (e>>3)*16][nt*16 + (lane&15)]
__global__ void pack_b_kernel(const float* __restrict__ src,
                              __bf16* __restrict__ dst, int K, int N) {
  const int tid = blockIdx.x * 256 + threadIdx.x;
  const int Nt = N >> 4;
  const int g = tid >> 9;            // group of 512 elements = one (kt,nt)
  const int kt = g / Nt;
  const int nt = g - kt * Nt;
  const int idx = tid & 511;
  const int lane = idx >> 4;
  const int e = idx & 15;
  const int k = kt * 32 + ((lane >> 4) << 3) + (e & 7) + ((e >> 3) << 4);
  const int n = (nt << 4) + (lane & 15);
  (void)K;
  dst[tid] = (__bf16)src[k * N + n];
}

// ---------------------------------------------------------------------------
// Fused main kernel: one workgroup (8 wave32) per 32 positions per batch
// ---------------------------------------------------------------------------
struct MainParams {
  const float* x;
  const float* bk;
  const float* bq;
  const float* cm_b1;
  const float* cm_W2;
  const float* cm_b2;
  const float* pk_b;
  const float* ta_b1;
  const float* ta_W2;
  const float* ta_b2;
  const float* bps;
  const __bf16* pWk;
  const __bf16* pWq;
  const __bf16* pW1a;
  const __bf16* pW1b;
  const __bf16* pW1c;
  const __bf16* pPk;
  const __bf16* pTa;
  const __bf16* pWps;
  float* out;
};

__launch_bounds__(256, 2)
__global__ void binding_main_kernel(MainParams P) {
  __shared__ __attribute__((aligned(16))) __bf16 xs[XROWS * XSTRIDE]; // 35904 B
  __shared__ float sraw[MTILE * NPATP];                               // 14336 B
  __shared__ float acc[4 * MTILE];  // k·q | h·W2 | Σpk² | th·W2  (32 each)

  const int tile = blockIdx.x;      // 128 tiles of 32 positions
  const int bb = blockIdx.y;        // batch
  const int p0 = tile * MTILE;

  if (threadIdx.x < 4 * MTILE) acc[threadIdx.x] = 0.f;

  // Stage x rows [p0-1 .. p0+32] -> LDS bf16 (zero-filled out of range)
  const float* xb = P.x + (size_t)bb * (TSEQ * DIM);
  for (int i = threadIdx.x; i < XROWS * 128; i += 256) {
    const int r = i >> 7;
    const int c4 = (i & 127) << 2;
    const int g = p0 - 1 + r;
    float4 v = make_float4(0.f, 0.f, 0.f, 0.f);
    if (g >= 0 && g < TSEQ) v = *(const float4*)(xb + (size_t)g * DIM + c4);
    union { __bf16 h[4]; uint2 u; } o;
    o.h[0] = (__bf16)v.x; o.h[1] = (__bf16)v.y;
    o.h[2] = (__bf16)v.z; o.h[3] = (__bf16)v.w;
    *(uint2*)&xs[r * XSTRIDE + c4] = o.u;
  }
  __syncthreads();

  const int wave = threadIdx.x >> 5;
  const int lane = threadIdx.x & 31;
  const int lh = lane >> 4;          // which half of the wave
  const int col = lane & 15;         // N within the 16-wide tile
  float* dotkq = acc;
  float* cmdot = acc + MTILE;
  float* sumsq = acc + 2 * MTILE;
  float* tadot = acc + 3 * MTILE;

  // 91 WMMA jobs round-robined over 8 waves; each job = 32 rows x 16 cols.
  // xs row offsets: 0 = left (x[p-1]), 1 = xl (x[p]), 2 = xr (x[p+1])
  for (int job = wave; job < 91; job += 8) {
    if (job < 16) {                                    // keys & queries tile
      const int nt = job;
      v8f ck0 = {}, ck1 = {}, cq0 = {}, cq1 = {};
      gemm32(xs, 1, P.pWk, 16, nt, lane, ck0, ck1);
      gemm32(xs, 2, P.pWq, 16, nt, lane, cq0, cq1);
      const int n = nt * 16 + col;
      const float bkv = P.bk[n], bqv = P.bq[n];
#pragma unroll
      for (int v = 0; v < 8; ++v) {
        atomicAdd(&dotkq[lh * 8 + v],      (ck0[v] + bkv) * (cq0[v] + bqv));
        atomicAdd(&dotkq[16 + lh * 8 + v], (ck1[v] + bkv) * (cq1[v] + bqv));
      }
    } else if (job < 48) {                             // context MLP hidden
      const int nt = job - 16;
      v8f c0 = {}, c1 = {};
      gemm32(xs, 0, P.pW1a, 32, nt, lane, c0, c1);
      gemm32(xs, 1, P.pW1b, 32, nt, lane, c0, c1);
      gemm32(xs, 2, P.pW1c, 32, nt, lane, c0, c1);
      const int n = nt * 16 + col;
      const float b1 = P.cm_b1[n], w2 = P.cm_W2[n];
#pragma unroll
      for (int v = 0; v < 8; ++v) {
        atomicAdd(&cmdot[lh * 8 + v],      fmaxf(c0[v] + b1, 0.f) * w2);
        atomicAdd(&cmdot[16 + lh * 8 + v], fmaxf(c1[v] + b1, 0.f) * w2);
      }
    } else if (job < 80) {                             // pattern-key sumsq
      const int nt = job - 48;
      v8f c0 = {}, c1 = {};
      gemm32(xs, 1, P.pPk, 32, nt, lane, c0, c1);
      const int n = nt * 16 + col;
      const float bp = P.pk_b[n];
#pragma unroll
      for (int v = 0; v < 8; ++v) {
        const float p0v = c0[v] + bp, p1v = c1[v] + bp;
        atomicAdd(&sumsq[lh * 8 + v],      p0v * p0v);
        atomicAdd(&sumsq[16 + lh * 8 + v], p1v * p1v);
      }
    } else if (job < 87) {                             // raw pattern scores
      const int nt = job - 80;
      v8f c0 = {}, c1 = {};
      gemm32(xs, 1, P.pWps, 7, nt, lane, c0, c1);
      const int n = nt * 16 + col;
      const float bv = P.bps[n];
#pragma unroll
      for (int v = 0; v < 8; ++v) {
        sraw[(lh * 8 + v) * NPATP + n]        = c0[v] + bv;
        sraw[(16 + lh * 8 + v) * NPATP + n]   = c1[v] + bv;
      }
    } else {                                           // threshold adapter
      const int nt = job - 87;
      v8f c0 = {}, c1 = {};
      gemm32(xs, 1, P.pTa, 4, nt, lane, c0, c1);
      const int n = nt * 16 + col;
      const float b1 = P.ta_b1[n], w2 = P.ta_W2[n];
#pragma unroll
      for (int v = 0; v < 8; ++v) {
        atomicAdd(&tadot[lh * 8 + v],      fmaxf(c0[v] + b1, 0.f) * w2);
        atomicAdd(&tadot[16 + lh * 8 + v], fmaxf(c1[v] + b1, 0.f) * w2);
      }
    }
  }
  __syncthreads();

  // Epilogue: 32 lanes, one per position
  if (threadIdx.x < MTILE) {
    const int m = threadIdx.x;
    const int p = p0 + m;
    if (p < NPOS) {
      const float bs = sigm(dotkq[m] * 0.0625f);                // /sqrt(256)
      const float cmv = sigm(cmdot[m] + P.cm_b2[0]);
      const float inv = 1.0f / fmaxf(sqrtf(sumsq[m]), 1e-12f);
      const float* sr = &sraw[m * NPATP];
      float zmax = -1e30f;
      for (int j = 0; j < NPAT; ++j)
        zmax = fmaxf(zmax, sr[j] * inv * 10.0f);
      float den = 0.f, num = 0.f;
      for (int j = 0; j < NPAT; ++j) {
        const float s = sr[j] * inv;
        const float e = __expf(s * 10.0f - zmax);
        den += e;
        num += e * s;
      }
      const float pb = sigm((num / den) * 5.0f);
      const float at = 0.5f + 0.1f * sigm(tadot[m] + P.ta_b2[0]);
      float eb = bs * cmv * (1.0f + 0.5f * pb);
      eb = fminf(fmaxf(eb, 0.f), 1.f);
      float* out = P.out;
      out[OFF_EB + bb * NPOS + p] = eb;
      out[OFF_AT + bb * NPOS + p] = at;
      out[OFF_BS + bb * NPOS + p] = bs;
      out[OFF_CM + bb * NPOS + p] = cmv;
      out[OFF_PB + bb * NPOS + p] = pb;
      out[bb * TSEQ + p + 1] = (eb > at) ? 1.0f : 0.0f;
    }
  } else if (threadIdx.x == MTILE && p0 == 0) {
    P.out[bb * TSEQ] = 0.f;   // binding_mask[:,0] = 0
  }
}

// ---------------------------------------------------------------------------
// Launch
// ---------------------------------------------------------------------------
extern "C" void kernel_launch(void* const* d_in, const int* in_sizes, int n_in,
                              void* d_out, int out_size, void* d_ws,
                              size_t ws_size, hipStream_t stream) {
  (void)in_sizes; (void)n_in; (void)out_size; (void)ws_size;
  const float* x      = (const float*)d_in[0];
  const float* Wk     = (const float*)d_in[1];
  const float* bk     = (const float*)d_in[2];
  const float* Wq     = (const float*)d_in[3];
  const float* bq     = (const float*)d_in[4];
  const float* cm_W1  = (const float*)d_in[5];
  const float* cm_b1  = (const float*)d_in[6];
  const float* cm_W2  = (const float*)d_in[7];
  const float* cm_b2  = (const float*)d_in[8];
  const float* pm     = (const float*)d_in[9];
  const float* pk_W   = (const float*)d_in[10];
  const float* pk_b   = (const float*)d_in[11];
  const float* ta_W1  = (const float*)d_in[12];
  const float* ta_b1  = (const float*)d_in[13];
  const float* ta_W2  = (const float*)d_in[14];
  const float* ta_b2  = (const float*)d_in[15];

  // workspace layout (f32 then packed bf16) ~3.1 MB
  float* Pn  = (float*)d_ws;             // 100*512
  float* bps = Pn + NPAT * DIM;          // 112
  float* Wps = bps + NPATP;              // 512*112
  __bf16* pWk  = (__bf16*)(Wps + DIM * NPATP);
  __bf16* pWq  = pWk  + 512 * 256;
  __bf16* pW1a = pWq  + 512 * 256;
  __bf16* pW1b = pW1a + 512 * 512;
  __bf16* pW1c = pW1b + 512 * 512;
  __bf16* pPk  = pW1c + 512 * 512;
  __bf16* pTa  = pPk  + 512 * 512;
  __bf16* pWps = pTa  + 512 * 64;

  norm_patterns_kernel<<<NPAT, 256, 0, stream>>>(pm, Pn);
  wps_kernel<<<(DIM * NPATP) / 256, 256, 0, stream>>>(pk_W, Pn, Wps);
  bps_kernel<<<1, 128, 0, stream>>>(pk_b, Pn, bps);

  pack_b_kernel<<<512,  256, 0, stream>>>(Wk, pWk, 512, 256);
  pack_b_kernel<<<512,  256, 0, stream>>>(Wq, pWq, 512, 256);
  pack_b_kernel<<<1024, 256, 0, stream>>>(cm_W1,              pW1a, 512, 512);
  pack_b_kernel<<<1024, 256, 0, stream>>>(cm_W1 + 512 * 512,  pW1b, 512, 512);
  pack_b_kernel<<<1024, 256, 0, stream>>>(cm_W1 + 1024 * 512, pW1c, 512, 512);
  pack_b_kernel<<<1024, 256, 0, stream>>>(pk_W, pPk, 512, 512);
  pack_b_kernel<<<128,  256, 0, stream>>>(ta_W1, pTa, 512, 64);
  pack_b_kernel<<<224,  256, 0, stream>>>(Wps, pWps, 512, NPATP);

  MainParams mp;
  mp.x = x; mp.bk = bk; mp.bq = bq; mp.cm_b1 = cm_b1; mp.cm_W2 = cm_W2;
  mp.cm_b2 = cm_b2; mp.pk_b = pk_b; mp.ta_b1 = ta_b1; mp.ta_W2 = ta_W2;
  mp.ta_b2 = ta_b2; mp.bps = bps;
  mp.pWk = pWk; mp.pWq = pWq; mp.pW1a = pW1a; mp.pW1b = pW1b; mp.pW1c = pW1c;
  mp.pPk = pPk; mp.pTa = pTa; mp.pWps = pWps;
  mp.out = (float*)d_out;

  dim3 grid(NPOS / MTILE + 1, BSZ);   // 128 tiles of 32 positions x 8 batches
  binding_main_kernel<<<grid, 256, 0, stream>>>(mp);
}
